// QuantizedLinearWrapper_45217415692989
// MI455X (gfx1250) — matrix operand order repaired and checked by reading, not compiled, not run on hardware
//
#include <hip/hip_runtime.h>

#define IN_FEATURES 4096
#define OUT_FEATURES 16384
#define SEQ 1024

typedef __attribute__((ext_vector_type(16))) _Float16 v16h;
typedef __attribute__((ext_vector_type(8)))  _Float16 v8h;
typedef __attribute__((ext_vector_type(8)))  float    v8f;
typedef __attribute__((ext_vector_type(4)))  float    v4f;
typedef __attribute__((ext_vector_type(4)))  int      v4i;

// ---------------- pre-pass: x f32 -> f16 (8 elements / thread) ----------------
__global__ __launch_bounds__(256)
void xcvt_kernel(const float* __restrict__ x, _Float16* __restrict__ xh) {
  const size_t i = ((size_t)blockIdx.x * 256 + threadIdx.x) * 8;
  v4f f0 = *reinterpret_cast<const v4f*>(x + i);
  v4f f1 = *reinterpret_cast<const v4f*>(x + i + 4);
  v8h h;
#pragma unroll
  for (int j = 0; j < 4; ++j) {
    h[j]     = (_Float16)f0[j];
    h[4 + j] = (_Float16)f1[j];
  }
  *reinterpret_cast<v8h*>(xh + i) = h;
}

// A fragment (16x32 f16, ISA 7.12.2): lane half h = lane>>4, row = lane&15.
// elements 0..7  <- K = k0 + 8h + (0..7)      (contiguous, 16B)
// elements 8..15 <- K = k0 + 16 + 8h + (0..7) (contiguous, 16B)
template <bool PRECONV>
__device__ __forceinline__ v16h load_a_tile(const void* __restrict__ base,
                                            size_t rowoff, int k0, int half) {
  v16h a;
  if constexpr (PRECONV) {
    const _Float16* row = reinterpret_cast<const _Float16*>(base) + rowoff;
    v8h lo = *reinterpret_cast<const v8h*>(row + k0 + half * 8);
    v8h hi = *reinterpret_cast<const v8h*>(row + k0 + 16 + half * 8);
#pragma unroll
    for (int i = 0; i < 8; ++i) { a[i] = lo[i]; a[8 + i] = hi[i]; }
  } else {
    const float* row = reinterpret_cast<const float*>(base) + rowoff;
    const v4f* p0 = reinterpret_cast<const v4f*>(row + k0 + half * 8);
    const v4f* p1 = reinterpret_cast<const v4f*>(row + k0 + 16 + half * 8);
    v4f f0 = p0[0], f1 = p0[1], f2 = p1[0], f3 = p1[1];
#pragma unroll
    for (int i = 0; i < 4; ++i) {
      a[i]      = (_Float16)f0[i];
      a[4 + i]  = (_Float16)f1[i];
      a[8 + i]  = (_Float16)f2[i];
      a[12 + i] = (_Float16)f3[i];
    }
  }
  return a;
}

// B fragment (32x16 f16): lane holds column N = lane&15;
// elements 0..7 <- K = k0 + 8*(lane>>4) + (0..7), elements 8..15 <- K = k0 + 16 + 8*(lane>>4) + (0..7)
__device__ __forceinline__ v16h load_b_tile(const int* __restrict__ wrow,
                                            int k0, int half) {
  const v4i* p0 = reinterpret_cast<const v4i*>(wrow + k0 + half * 8);
  const v4i* p1 = reinterpret_cast<const v4i*>(wrow + k0 + 16 + half * 8);
  v4i w0 = p0[0], w1 = p0[1], w2 = p1[0], w3 = p1[1];
  v16h b;
#pragma unroll
  for (int i = 0; i < 4; ++i) {
    b[i]      = (_Float16)(float)w0[i];
    b[4 + i]  = (_Float16)(float)w1[i];
    b[8 + i]  = (_Float16)(float)w2[i];
    b[12 + i] = (_Float16)(float)w3[i];
  }
  return b;
}

// ---------------- main GEMM: per-wave 64(M) x 32(N), 8 WMMAs / k-step ----------
template <bool PRECONV>
__global__ __launch_bounds__(256)
void qlinear_wmma(const void* __restrict__ xa,
                  const int* __restrict__ qw,
                  const float* __restrict__ scales,
                  float* __restrict__ out) {
  const int lane = threadIdx.x & 31;
  const int wave = threadIdx.x >> 5;
  const int wm   = wave & 1;   // 2 waves along M
  const int wn   = wave >> 1;  // 4 waves along N
  const int half = lane >> 4;
  const int r    = lane & 15;

  // grid.x = M tiles so co-scheduled blocks share the same weight band in L2
  const int m0 = blockIdx.x * 128 + wm * 64;  // wave M base (4 subtiles of 16)
  const int n0 = blockIdx.y * 128 + wn * 32;  // wave N base (2 subtiles of 16)

  size_t arow[4];
#pragma unroll
  for (int i = 0; i < 4; ++i)
    arow[i] = (size_t)(m0 + 16 * i + r) * IN_FEATURES;

  const int* wrow[2];
#pragma unroll
  for (int j = 0; j < 2; ++j)
    wrow[j] = qw + (size_t)(n0 + 16 * j + r) * IN_FEATURES;

  v8f acc[4][2] = {};

  for (int k0 = 0; k0 < IN_FEATURES; k0 += 32) {
    if (k0 + 64 <= IN_FEATURES) {
      __builtin_prefetch(wrow[0] + k0 + 32, 0, 0);  // global_prefetch_b8
      __builtin_prefetch(wrow[1] + k0 + 32, 0, 0);
    }
    v16h a[4];
#pragma unroll
    for (int i = 0; i < 4; ++i)
      a[i] = load_a_tile<PRECONV>(xa, arow[i], k0, half);

    v16h b[2];
#pragma unroll
    for (int j = 0; j < 2; ++j)
      b[j] = load_b_tile(wrow[j], k0, half);

#pragma unroll
    for (int i = 0; i < 4; ++i)
#pragma unroll
      for (int j = 0; j < 2; ++j)
        acc[i][j] = __builtin_amdgcn_wmma_f32_16x16x32_f16(
            false, a[i], false, b[j], (short)0, acc[i][j], false, false);
  }

  // C/D layout: lane holds N = lane&15, VGPR rr holds M = rr + 8*(lane>>4)
  const float s0 = scales[n0 + r];
  const float s1 = scales[n0 + 16 + r];

#pragma unroll
  for (int i = 0; i < 4; ++i) {
#pragma unroll
    for (int rr = 0; rr < 8; ++rr) {
      const size_t row = (size_t)(m0 + 16 * i + 8 * half + rr) * OUT_FEATURES;
      out[row + n0 + r]      = acc[i][0][rr] * s0;
      out[row + n0 + 16 + r] = acc[i][1][rr] * s1;
    }
  }
}

extern "C" void kernel_launch(void* const* d_in, const int* in_sizes, int n_in,
                              void* d_out, int out_size, void* d_ws, size_t ws_size,
                              hipStream_t stream) {
  const float* x      = (const float*)d_in[0];
  const int*   qw     = (const int*)d_in[1];
  const float* scales = (const float*)d_in[2];
  float*       out    = (float*)d_out;

  dim3 grid(SEQ / 128, OUT_FEATURES / 128);  // (8, 128)
  dim3 block(256);

  const size_t xh_bytes = (size_t)SEQ * IN_FEATURES * sizeof(_Float16);  // 8 MB
  if (ws_size >= xh_bytes) {
    // pre-convert activations once, then run conversion-free A path
    xcvt_kernel<<<(SEQ * IN_FEATURES) / (256 * 8), 256, 0, stream>>>(
        x, (_Float16*)d_ws);
    qlinear_wmma<true><<<grid, block, 0, stream>>>(d_ws, qw, scales, out);
  } else {
    qlinear_wmma<false><<<grid, block, 0, stream>>>(x, qw, scales, out);
  }
}
